// Binarized_Convolution1D_51943334478205
// MI455X (gfx1250) — matrix operand order repaired and checked by reading, not compiled, not run on hardware
//
#include <hip/hip_runtime.h>

typedef __attribute__((ext_vector_type(16))) _Float16 v16h;
typedef __attribute__((ext_vector_type(8)))  _Float16 v8h;
typedef __attribute__((ext_vector_type(4)))  _Float16 v4h;
typedef __attribute__((ext_vector_type(8)))  float    v8f;

#define BATCH  64
#define LEN    4096
#define CIN    256
#define COUT   256
#define KW     5
#define PAD    2

#define M_TILE 64
#define N_TILE 256                  /* full Cout in one pass: x read once from HBM */
#define XS_ROWS (M_TILE + KW - 1)   /* 68 rows incl. halo */
#define XS_STRIDE (CIN + 8)         /* 264 halves: 528B row stride breaks LDS bank conflicts */

// ---------------------------------------------------------------------------
// Prep: binarize W (fp32 -> +-1 f16) and pack into WMMA B-fragment order:
//   Wt[ ((k*8+kc)*2+h) * (COUT*16) + co*16 + e ]  with ci = kc*32 + h*8 + (e&7) + (e>>3)*16
// so each lane's 16-f16 B fragment is one contiguous, 32B-aligned chunk.
// Total: 5*256*256*2B = 640KB in d_ws (L2-resident).
// ---------------------------------------------------------------------------
__global__ __launch_bounds__(256) void binarize_pack_w(
    const float* __restrict__ W, _Float16* __restrict__ Wt) {
  int idx = blockIdx.x * 256 + threadIdx.x;   // 0 .. 327679
  int e   = idx & 15;
  int co  = (idx >> 4) & 255;
  int blk = idx >> 12;                        // 0 .. 79
  int h   = blk & 1;
  int kc  = (blk >> 1) & 7;
  int k   = blk >> 4;
  int ci  = kc * 32 + h * 8 + (e & 7) + (e >> 3) * 16;
  float w = W[(k * CIN + ci) * COUT + co];
  Wt[idx] = (_Float16)(w >= 0.0f ? 1.0f : -1.0f);
}

// ---------------------------------------------------------------------------
// Main: 1D binarized conv as 5 shifted GEMMs fused via WMMA accumulation.
// Block = 256 threads (8 waves, 2(M) x 4(N)). Output tile 64(L) x 256(Cout).
// Wave computes 32 rows x 64 cols = 2x4 x v8f accumulators; every B fragment
// feeds 2 WMMAs (M-reuse) -> 1 global_load_b128 + 0.5 ds_load per WMMA.
// 320 v_wmma_f32_16x16x32_f16 per wave.
// ---------------------------------------------------------------------------
__global__ __launch_bounds__(256) void bconv1d_wmma(
    const float* __restrict__ x, const _Float16* __restrict__ Wt,
    const float* __restrict__ bias, float* __restrict__ out) {
  __shared__ _Float16 xs[XS_ROWS * XS_STRIDE];   // ~35.9 KB

  const int rowTile = blockIdx.x;
  const int batch   = rowTile >> 6;              // / (LEN/M_TILE)
  const int l0      = (rowTile & 63) * M_TILE;

  const int tid = threadIdx.x;

  // ---- Stage x tile (fp32 -> f16) into LDS, zero-padding the L boundary ----
  const float* xb = x + (size_t)batch * LEN * CIN;
  #pragma unroll
  for (int it = 0; it < (XS_ROWS * (CIN / 4)) / 256; ++it) {   // 17 iters
    int idx = it * 256 + tid;
    int r   = idx >> 6;            // LDS row (0..67)
    int ci  = (idx & 63) << 2;     // 4 channels per thread
    int gl  = l0 - PAD + r;        // global L index
    float4 v = {0.f, 0.f, 0.f, 0.f};
    if (gl >= 0 && gl < LEN)
      v = *(const float4*)(xb + (size_t)gl * CIN + ci);
    v4h hv;
    hv.x = (_Float16)v.x; hv.y = (_Float16)v.y;
    hv.z = (_Float16)v.z; hv.w = (_Float16)v.w;
    *(v4h*)(&xs[r * XS_STRIDE + ci]) = hv;
  }
  __syncthreads();

  const int lane = tid & 31;
  const int wave = tid >> 5;
  const int wm0  = (wave & 1) * 32;      // wave M group: 32 rows
  const int wn0  = (wave >> 1) * 64;     // wave N group: 64 cols
  const int h    = lane >> 4;            // lane half (WMMA layout)
  const int ln   = lane & 15;

  v8f acc[2][4] = {};

  #pragma unroll
  for (int k = 0; k < KW; ++k) {
    // Tap shift = +k LDS row offset (halo makes all 5 taps in-bounds)
    const _Float16* xrow0 = &xs[(wm0 +  0 + ln + k) * XS_STRIDE];
    const _Float16* xrow1 = &xs[(wm0 + 16 + ln + k) * XS_STRIDE];
    #pragma unroll
    for (int kc = 0; kc < 8; ++kc) {
      // A fragments (16x32 f16): elems 0-7 = ci base..+7, 8-15 = base+16..+23
      const int ab = kc * 32 + h * 8;
      v8h a0lo = *(const v8h*)(xrow0 + ab);        // ds_load_b128
      v8h a0hi = *(const v8h*)(xrow0 + ab + 16);   // ds_load_b128
      v8h a1lo = *(const v8h*)(xrow1 + ab);
      v8h a1hi = *(const v8h*)(xrow1 + ab + 16);
      v16h aA, aB;
      #pragma unroll
      for (int e = 0; e < 8; ++e) {
        aA[e] = a0lo[e]; aA[e + 8] = a0hi[e];
        aB[e] = a1lo[e]; aB[e + 8] = a1hi[e];
      }

      const int blk = (k * 8 + kc) * 2 + h;
      const _Float16* bp = Wt + (size_t)blk * (COUT * 16)
                              + (size_t)(wn0 + ln) * 16;
      #pragma unroll
      for (int sub = 0; sub < 4; ++sub) {
        v16h bfrag = *(const v16h*)(bp + sub * 256);   // 32B coalesced, L2-hot
        acc[0][sub] = __builtin_amdgcn_wmma_f32_16x16x32_f16(
            false, aA, false, bfrag, (short)0, acc[0][sub], false, false);
        acc[1][sub] = __builtin_amdgcn_wmma_f32_16x16x32_f16(
            false, aB, false, bfrag, (short)0, acc[1][sub], false, false);
      }
    }
  }

  // ---- Epilogue: bias + ReLU, coalesced fp32 stores ----
  #pragma unroll
  for (int ms = 0; ms < 2; ++ms) {
    #pragma unroll
    for (int sub = 0; sub < 4; ++sub) {
      int   co = wn0 + sub * 16 + ln;
      float bv = bias[co];
      int   m0 = wm0 + ms * 16 + h * 8;               // C/D layout: M = r + 8*h
      float* op = out + ((size_t)(batch * LEN + l0 + m0) * COUT) + co;
      #pragma unroll
      for (int r = 0; r < 8; ++r) {
        float v = acc[ms][sub][r] + bv;
        op[(size_t)r * COUT] = v > 0.f ? v : 0.f;
      }
    }
  }
}

extern "C" void kernel_launch(void* const* d_in, const int* in_sizes, int n_in,
                              void* d_out, int out_size, void* d_ws, size_t ws_size,
                              hipStream_t stream) {
  const float* x = (const float*)d_in[0];
  const float* W = (const float*)d_in[1];
  const float* b = (const float*)d_in[2];
  float*     out = (float*)d_out;
  _Float16*   Wt = (_Float16*)d_ws;   // 640 KB packed binarized weights

  binarize_pack_w<<<(KW * CIN * COUT) / 256, 256, 0, stream>>>(W, Wt);

  dim3 grid(BATCH * LEN / M_TILE);   // 4096 blocks, single Cout pass
  bconv1d_wmma<<<grid, 256, 0, stream>>>(x, Wt, b, out);
}
